// Dwtpool_61881888801141
// MI455X (gfx1250) — matrix operand order repaired and checked by reading, not compiled, not run on hardware
//
#include <hip/hip_runtime.h>

// ---------------------------------------------------------------------------
// MI455X (gfx1250) implementation.
// Roofline: ~1.28 TFLOP of conv work vs ~0.6GB of data -> compute-bound.
// All heavy convs run as implicit GEMMs on v_wmma_f32_16x16x32_bf16 (wave32),
// activations NHWC bf16, weights pre-packed into the exact WMMA B-fragment
// VGPR layout (ISA 7.12.2). Main conv kernel double-buffers the input
// row-window in LDS using GLOBAL_LOAD_ASYNC_TO_LDS_B128 (ASYNCcnt-tracked
// DMA, no VGPR round-trip), overlapping the copy of stage s+1 with the
// 8-WMMA inner loop of stage s; weight fragments stream from global with
// global_prefetch_b8. The 77-GFLOP channel-conv is algebraically contracted
// with the softmax vector (S-factorization), and the channel gate cw is
// folded into per-batch proj weights so the epilogue is one more WMMA GEMM.
// ---------------------------------------------------------------------------

#define EPSV 1e-5f

typedef __bf16 bf16_t;
typedef bf16_t v16bf __attribute__((ext_vector_type(16)));
typedef float  v8f   __attribute__((ext_vector_type(8)));
typedef unsigned int v4u __attribute__((ext_vector_type(4)));

union AFrag { v16bf v; v4u u[2]; };

__device__ __forceinline__ unsigned short f2bf(float f) {
  unsigned u = __float_as_uint(f);
  unsigned r = u + 0x7FFFu + ((u >> 16) & 1u);   // round-to-nearest-even
  return (unsigned short)(r >> 16);
}
__device__ __forceinline__ float bf2f(unsigned short h) {
  return __uint_as_float(((unsigned)h) << 16);
}

// ---------------------------------------------------------------------------
// Cooperative LDS-staged implicit-GEMM conv (64x64 spatial, Cout=256).
// Block = (batch, row y): 8 waves x 32co each = 256 co, 64 px.
// Stages (kh, channel-chunk) rows into a double-buffered LDS window via
// async global->LDS DMA; all 8 waves then read A-fragments from LDS for all
// kw taps. Fragment layouts per CDNA5 ISA 7.12.2:
//   A lane L: m=L&15, half=L>>4; e[0..7]=K(8*half+0..7), e[8..15]=K(+16)
//   B lane L: n=L&15, half=L>>4; e[0..7]=B[8*half+0..7][n], e[8..15]=B[16+8*half+0..7][n]
//   D lane L: col n=L&15; acc[v] = row (v + 8*half)
// ---------------------------------------------------------------------------
template<int KS, int CIN>
__global__ __launch_bounds__(256) void conv_lds(
    const unsigned short* __restrict__ in, int Ctot, int cio,
    const unsigned short* __restrict__ wf, long wfBatchStride,
    unsigned short* __restrict__ outb, int obCs, int obCo,
    float* __restrict__ outf, int ofMode)              // 0 none, 1 NHWC, 2 NCHW
{
  constexpr int PAD = KS / 2;
  constexpr int CHS = (CIN > 384) ? 384 : CIN;   // channels per LDS stage
  constexpr int NST = CIN / CHS;                 // channel stages per tap-row
  constexpr int KC  = CHS >> 5;                  // k-chunks per stage
  constexpr int KCT = CIN >> 5;                  // k-chunks per tap (total)
  constexpr int PITCH = CHS + 8;                 // shorts per pixel (+16B skew)
  constexpr int WPX = 64 + 2 * PAD;              // staged pixels per row
  constexpr int CPX = CHS >> 3;                  // 16B chunks per pixel
  constexpr int nsteps = KS * KS * KCT;
  constexpr int BUFS = WPX * PITCH;              // shorts per LDS buffer
  constexpr int NSTAGES = KS * NST;

  __shared__ __align__(16) unsigned short smem[2 * BUFS];

  const int tid  = threadIdx.x;
  const int lane = tid & 31;
  const int np   = tid >> 5;                     // wave id = co-pair id (0..7)
  const int b    = blockIdx.x >> 6;
  const int y    = blockIdx.x & 63;
  const int m = lane & 15, half = lane >> 4;

  const unsigned short* wfb = wf + (size_t)b * (size_t)wfBatchStride;
  const unsigned short* wt0 = wfb + ((size_t)((np * 2 + 0) * nsteps) * 32 + lane) * 16;
  const unsigned short* wt1 = wfb + ((size_t)((np * 2 + 1) * nsteps) * 32 + lane) * 16;

  const v8f z8 = {0.f,0.f,0.f,0.f,0.f,0.f,0.f,0.f};
  const v4u z4 = {0u,0u,0u,0u};
  v8f acc[4][2];
  #pragma unroll
  for (int g = 0; g < 4; ++g) { acc[g][0] = z8; acc[g][1] = z8; }

  // Issue the async DMA for stage s into LDS buffer `buf`.
  // Pad columns / fully out-of-range rows are zero-filled with plain LDS
  // stores (the prior barrier guarantees no wave still reads this buffer).
  auto issue_stage = [&](int s, int buf) {
    const int kh = s / NST, cs = s - kh * NST;
    const int ys = y + kh - PAD;
    const bool yin = ((unsigned)ys < 64u);
    unsigned short* dstb = &smem[buf * BUFS];
    if (yin) {
      const unsigned short* src =
          in + ((size_t)(b * 4096 + ys * 64) * Ctot + cio + cs * CHS);
      for (int idx = tid; idx < 64 * CPX; idx += 256) {
        int px = idx / CPX, q = idx - px * CPX;
        unsigned ldsa = (unsigned)(size_t)(&dstb[(px + PAD) * PITCH + q * 8]);
        unsigned long long ga =
            (unsigned long long)(size_t)(src + (size_t)px * Ctot + q * 8);
        asm volatile("global_load_async_to_lds_b128 %0, %1, off"
                     :: "v"(ldsa), "v"(ga) : "memory");
      }
      if (PAD > 0) {
        for (int idx = tid; idx < 2 * PAD * CPX; idx += 256) {
          int side = idx / (PAD * CPX);
          int r = idx - side * (PAD * CPX);
          int px = (side ? 64 + PAD : 0) + r / CPX;
          int q = r - (r / CPX) * CPX;
          *(v4u*)(&dstb[px * PITCH + q * 8]) = z4;
        }
      }
    } else {
      for (int idx = tid; idx < WPX * CPX; idx += 256) {
        int px = idx / CPX, q = idx - px * CPX;
        *(v4u*)(&dstb[px * PITCH + q * 8]) = z4;
      }
    }
  };

  issue_stage(0, 0);
  asm volatile("s_wait_asynccnt 0x0" ::: "memory");
  __syncthreads();

  #pragma unroll 1
  for (int s = 0; s < NSTAGES; ++s) {
    const int cur = s & 1;
    if (s + 1 < NSTAGES) issue_stage(s + 1, cur ^ 1);   // overlap next DMA
    const int kh = s / NST, cs = s - kh * NST;
    const unsigned short* bufc = &smem[cur * BUFS];
    #pragma unroll 1
    for (int kw = 0; kw < KS; ++kw) {
      const int stepb = (kh * KS + kw) * KCT + cs * KC;
      #pragma unroll 2
      for (int kc = 0; kc < KC; ++kc) {
        const unsigned short* q0 = wt0 + (size_t)(stepb + kc) * 512;
        const unsigned short* q1 = wt1 + (size_t)(stepb + kc) * 512;
        AFrag b0, b1;
        b0.u[0] = *(const v4u*)(q0); b0.u[1] = *(const v4u*)(q0 + 8);
        b1.u[0] = *(const v4u*)(q1); b1.u[1] = *(const v4u*)(q1 + 8);
        __builtin_prefetch(q0 + 4 * 512, 0, 1);   // global_prefetch_b8
        __builtin_prefetch(q1 + 4 * 512, 0, 1);
        AFrag a[4];
        #pragma unroll
        for (int g = 0; g < 4; ++g) {
          const unsigned short* ap = &bufc[(g * 16 + m + kw) * PITCH + kc * 32 + 8 * half];
          a[g].u[0] = *(const v4u*)(ap);
          a[g].u[1] = *(const v4u*)(ap + 16);
        }
        #pragma unroll
        for (int g = 0; g < 4; ++g) {
          acc[g][0] = __builtin_amdgcn_wmma_f32_16x16x32_bf16(false, a[g].v, false, b0.v, (short)0, acc[g][0], false, false);
          acc[g][1] = __builtin_amdgcn_wmma_f32_16x16x32_bf16(false, a[g].v, false, b1.v, (short)0, acc[g][1], false, false);
        }
      }
    }
    asm volatile("s_wait_asynccnt 0x0" ::: "memory");   // my DMA into next buf done
    __syncthreads();                                    // everyone done with cur buf
  }

  // ---- epilogue: D rows are pixels, cols are co ----
  #pragma unroll
  for (int g = 0; g < 4; ++g) {
    #pragma unroll
    for (int j = 0; j < 2; ++j) {
      const int co = np * 32 + 16 * j + (lane & 15);
      #pragma unroll
      for (int v = 0; v < 8; ++v) {
        const int px = g * 16 + v + 8 * half;
        const int p  = y * 64 + px;
        float val = acc[g][j][v];
        if (outb) outb[(size_t)(b * 4096 + p) * obCs + obCo + co] = f2bf(val);
        if (outf) {
          if (ofMode == 1) outf[(size_t)(b * 4096 + p) * 256 + co] = val;
          else             outf[((size_t)b * 256 + co) * 4096 + p] = val;
        }
      }
    }
  }
}

// ---------------------------------------------------------------------------
// Simple global-load WMMA conv (used only for the 128x128 reduce 1x1 conv).
// ---------------------------------------------------------------------------
template<int KS, int SPW>
__global__ __launch_bounds__(256) void conv_wmma(
    const unsigned short* __restrict__ in, int Ctot, int cio, int Cin,
    const unsigned short* __restrict__ wf, long wfBatchStride, int Cout,
    unsigned short* __restrict__ outb, int obCs, int obCo,
    float* __restrict__ outf, int ofMode,
    const float* __restrict__ alpha, const float* __restrict__ beta)
{
  constexpr int PAD = KS / 2;
  constexpr int NP  = SPW * SPW;
  constexpr int PPR = SPW / 32;
  const int lane = threadIdx.x & 31;
  const int wid  = (blockIdx.x * blockDim.x + threadIdx.x) >> 5;
  const int Npairs     = Cout >> 5;
  const int MpairsPerB = NP >> 5;
  if (wid >= 16 * MpairsPerB * Npairs) return;

  const int np  = wid % Npairs;
  const int mp  = wid / Npairs;
  const int b   = mp / MpairsPerB;
  const int rem = mp % MpairsPerB;
  const int y   = rem / PPR;
  const int x0  = (rem % PPR) * 32;
  const int m = lane & 15, half = lane >> 4;
  const int KC = Cin >> 5;
  const int nsteps = KS * KS * KC;

  const unsigned short* wfb = wf + (size_t)b * (size_t)wfBatchStride;
  const unsigned short* wt0 = wfb + ((size_t)((np * 2 + 0) * nsteps) * 32 + lane) * 16;
  const unsigned short* wt1 = wfb + ((size_t)((np * 2 + 1) * nsteps) * 32 + lane) * 16;

  const v8f z8 = {0.f,0.f,0.f,0.f,0.f,0.f,0.f,0.f};
  const v4u z4 = {0u,0u,0u,0u};
  v8f acc[2][2];
  acc[0][0] = z8; acc[0][1] = z8; acc[1][0] = z8; acc[1][1] = z8;

  for (int kh = 0; kh < KS; ++kh) {
    const int ys = y + kh - PAD;
    const bool yin = ((unsigned)ys < (unsigned)SPW);
    for (int kw = 0; kw < KS; ++kw) {
      const int xs0 = x0 + m + kw - PAD;
      const int xs1 = xs0 + 16;
      const bool in0 = yin && ((unsigned)xs0 < (unsigned)SPW);
      const bool in1 = yin && ((unsigned)xs1 < (unsigned)SPW);
      const unsigned short* p0 = in + ((size_t)(b * NP + ys * SPW + xs0) * Ctot + cio + 8 * half);
      const unsigned short* p1 = in + ((size_t)(b * NP + ys * SPW + xs1) * Ctot + cio + 8 * half);
      const int stepbase = (kh * KS + kw) * KC;
      #pragma unroll 4
      for (int kc = 0; kc < KC; ++kc) {
        AFrag a0, a1;
        if (in0) { a0.u[0] = *(const v4u*)(p0 + kc * 32); a0.u[1] = *(const v4u*)(p0 + kc * 32 + 16); }
        else     { a0.u[0] = z4; a0.u[1] = z4; }
        if (in1) { a1.u[0] = *(const v4u*)(p1 + kc * 32); a1.u[1] = *(const v4u*)(p1 + kc * 32 + 16); }
        else     { a1.u[0] = z4; a1.u[1] = z4; }
        const unsigned short* q0 = wt0 + (size_t)(stepbase + kc) * 512;
        const unsigned short* q1 = wt1 + (size_t)(stepbase + kc) * 512;
        AFrag b0f, b1f;
        b0f.u[0] = *(const v4u*)(q0); b0f.u[1] = *(const v4u*)(q0 + 8);
        b1f.u[0] = *(const v4u*)(q1); b1f.u[1] = *(const v4u*)(q1 + 8);
        acc[0][0] = __builtin_amdgcn_wmma_f32_16x16x32_bf16(false, a0.v, false, b0f.v, (short)0, acc[0][0], false, false);
        acc[0][1] = __builtin_amdgcn_wmma_f32_16x16x32_bf16(false, a0.v, false, b1f.v, (short)0, acc[0][1], false, false);
        acc[1][0] = __builtin_amdgcn_wmma_f32_16x16x32_bf16(false, a1.v, false, b0f.v, (short)0, acc[1][0], false, false);
        acc[1][1] = __builtin_amdgcn_wmma_f32_16x16x32_bf16(false, a1.v, false, b1f.v, (short)0, acc[1][1], false, false);
      }
    }
  }

  #pragma unroll
  for (int i = 0; i < 2; ++i) {
    #pragma unroll
    for (int j = 0; j < 2; ++j) {
      const int co = np * 32 + 16 * j + (lane & 15);
      const float al = alpha ? alpha[co] : 1.f;
      const float be = alpha ? beta[co]  : 0.f;
      #pragma unroll
      for (int v = 0; v < 8; ++v) {
        const int px = x0 + 16 * i + v + 8 * half;
        const int p  = y * SPW + px;
        float val = acc[i][j][v];
        if (alpha) val = fmaxf(val * al + be, 0.f);
        if (outb) outb[(size_t)(b * NP + p) * obCs + obCo + co] = f2bf(val);
        if (outf) {
          if (ofMode == 1) outf[(size_t)(b * NP + p) * Cout + co] = val;
          else             outf[((size_t)b * Cout + co) * NP + p] = val;
        }
      }
    }
  }
}

// --- pack OIHW fp32 weights into WMMA B-fragment order (bf16) ---------------
__global__ void pack_w(const float* __restrict__ W, unsigned short* __restrict__ wf,
                       int Cout, int Cin, int KS, int total) {
  int t = blockIdx.x * blockDim.x + threadIdx.x;
  if (t >= total) return;
  const int KC = Cin >> 5, nst = KS * KS * KC;
  const int lane = t & 31;
  const int s    = (t >> 5) % nst;
  const int tile = (t >> 5) / nst;
  const int half = lane >> 4;
  const int co   = tile * 16 + (lane & 15);
  const int khkw = s / KC, kc = s % KC;
  unsigned short* dst = wf + (size_t)t * 16;
  for (int j = 0; j < 16; ++j) {
    int ci = kc * 32 + (j < 8 ? 8 * half + j : 16 + 8 * half + (j - 8));
    dst[j] = f2bf(W[((size_t)co * Cin + ci) * (KS * KS) + khkw]);
  }
}

// proj weights folded with per-batch channel gate cw
__global__ void pack_proj_k(const float* __restrict__ W, const float* __restrict__ cw,
                            unsigned short* __restrict__ dst) {
  int t = blockIdx.x * 256 + threadIdx.x;
  if (t >= 16 * 4096) return;
  const int b = t >> 12, r = t & 4095;
  const int lane = r & 31;
  const int sidx = (r >> 5) & 7;
  const int tile = r >> 8;
  const int half = lane >> 4, co = tile * 16 + (lane & 15);
  unsigned short* d = dst + (size_t)t * 16;
  const float* cwb = cw + b * 256;
  for (int j = 0; j < 16; ++j) {
    int ci = sidx * 32 + (j < 8 ? 8 * half + j : 16 + 8 * half + (j - 8));
    d[j] = f2bf(W[co * 256 + ci] * cwb[ci]);
  }
}

// --- BN folded into per-channel alpha/beta ---------------------------------
__global__ void bn_coef_k(const float* g, const float* bb, const float* mean,
                          const float* var, const float* rb, float* ab) {
  int i = threadIdx.x;
  if (i >= 64) return;
  float sc = g[i] * rsqrtf(var[i] + EPSV);
  ab[i]      = sc;
  ab[64 + i] = (rb[i] - mean[i]) * sc + bb[i];
}

// --- x NCHW f32 -> NHWC bf16 (LDS tiled transpose) -------------------------
__global__ __launch_bounds__(256) void x_nhwc_k(const float* __restrict__ x,
                                                unsigned short* __restrict__ xn) {
  __shared__ float tile[32][33];
  const int bid = blockIdx.x;               // 16 * 8 * 512
  const int pt = bid & 511, ct = (bid >> 9) & 7, b = bid >> 12;
  const int c0 = ct * 32, p0 = pt * 32;
  const int tid = threadIdx.x;
  const int pp = tid & 31, cc = tid >> 5;
  for (int rr = 0; rr < 4; ++rr) {
    int c = cc + rr * 8;
    tile[c][pp] = x[((size_t)(b * 256 + c0 + c)) * 16384 + p0 + pp];
  }
  __syncthreads();
  const int cc2 = tid & 31, pp2 = tid >> 5;
  for (int rr = 0; rr < 4; ++rr) {
    int p = pp2 + rr * 8;
    xn[((size_t)(b * 16384 + p0 + p)) * 256 + c0 + cc2] = f2bf(tile[cc2][p]);
  }
}

// --- 2x2 maxpool -> cat channels [1280,1536) bf16 --------------------------
__global__ void maxpool_k(const float* __restrict__ x, unsigned short* __restrict__ cat) {
  int t = blockIdx.x * 256 + threadIdx.x;       // 16*4096*256
  int c = t & 255, p2 = (t >> 8) & 4095, b = t >> 20;
  int y2 = p2 >> 6, x2 = p2 & 63;
  const float* xb = x + ((size_t)(b * 256 + c)) * 16384;
  int i0 = (2 * y2) * 128 + 2 * x2;
  float mv = fmaxf(fmaxf(xb[i0], xb[i0 + 1]), fmaxf(xb[i0 + 128], xb[i0 + 129]));
  cat[(size_t)(b * 4096 + p2) * 1536 + 1280 + c] = f2bf(mv);
}

// --- Haar DWT of r (NHWC bf16, 128x128x64) -> cat channels [0,256) ---------
__global__ void dwt_k(const unsigned short* __restrict__ r, unsigned short* __restrict__ cat) {
  int t = blockIdx.x * 256 + threadIdx.x;       // 16*4096*64
  int c = t & 63, p2 = (t >> 6) & 4095, b = t >> 18;
  int y2 = p2 >> 6, x2 = p2 & 63;
  const unsigned short* rb = r + (size_t)b * 16384 * 64;
  int y = 2 * y2, xx = 2 * x2;
  float a  = bf2f(rb[(size_t)(y * 128 + xx) * 64 + c]);
  float bb = bf2f(rb[(size_t)(y * 128 + xx + 1) * 64 + c]);
  float cc = bf2f(rb[(size_t)((y + 1) * 128 + xx) * 64 + c]);
  float dd = bf2f(rb[(size_t)((y + 1) * 128 + xx + 1) * 64 + c]);
  size_t base = (size_t)(b * 4096 + p2) * 1536;
  cat[base + c]       = f2bf((a + bb + cc + dd) * 0.5f);
  cat[base + 64 + c]  = f2bf((a + bb - cc - dd) * 0.5f);
  cat[base + 128 + c] = f2bf((a - bb + cc - dd) * 0.5f);
  cat[base + 192 + c] = f2bf((a - bb - cc + dd) * 0.5f);
}

// --- content logits: per-pixel 256-dot with conv_w (one wave per pixel) ----
__global__ __launch_bounds__(256) void content_k(const float* __restrict__ qkv,
                                                 const float* __restrict__ w,
                                                 float* __restrict__ logits) {
  int gt = blockIdx.x * blockDim.x + threadIdx.x;
  int wid = gt >> 5, lane = gt & 31;
  if (wid >= 16 * 4096) return;
  const float* row = qkv + (size_t)wid * 256 + lane * 8;
  const float* wr  = w + lane * 8;
  float s = 0.f;
  #pragma unroll
  for (int k = 0; k < 8; ++k) s += row[k] * wr[k];
  #pragma unroll
  for (int off = 16; off; off >>= 1) s += __shfl_xor(s, off, 32);
  if (lane == 0) logits[wid] = s;
}

__global__ void softmax_k(float* __restrict__ v) {    // one block per batch
  __shared__ float red[256];
  const int b = blockIdx.x, tid = threadIdx.x;
  float* p = v + (size_t)b * 4096;
  float mx = -3.4e38f;
  for (int i = tid; i < 4096; i += 256) mx = fmaxf(mx, p[i]);
  red[tid] = mx; __syncthreads();
  for (int s = 128; s > 0; s >>= 1) { if (tid < s) red[tid] = fmaxf(red[tid], red[tid + s]); __syncthreads(); }
  mx = red[0]; __syncthreads();
  float sm = 0.f;
  for (int i = tid; i < 4096; i += 256) sm += expf(p[i] - mx);
  red[tid] = sm; __syncthreads();
  for (int s = 128; s > 0; s >>= 1) { if (tid < s) red[tid] += red[tid + s]; __syncthreads(); }
  float inv = 1.f / red[0];
  __syncthreads();
  for (int i = tid; i < 4096; i += 256) p[i] = expf(p[i] - mx) * inv;
}

// --- S[b][tap][ci] = sum_p content[b,p] * qkv[b, ci, shifted p] ------------
__global__ void s_k(const float* __restrict__ qkv, const float* __restrict__ content,
                    float* __restrict__ S) {
  const int b = blockIdx.x / 9, tap = blockIdx.x % 9;
  const int kh = tap / 3, kw = tap % 3, c = threadIdx.x;
  float acc = 0.f;
  for (int p = 0; p < 4096; ++p) {
    int y = p >> 6, x = p & 63;
    int ys = y + kh - 1, xs = x + kw - 1;
    if ((unsigned)ys < 64u && (unsigned)xs < 64u)
      acc += content[b * 4096 + p] * qkv[((size_t)(b * 4096 + ys * 64 + xs)) * 256 + c];
  }
  S[(b * 9 + tap) * 256 + c] = acc;
}

// --- pooled[b][co] = sum_{ci,tap} chw[co][ci][tap] * S[b][tap][ci] ---------
__global__ void pooled_k(const float* __restrict__ chw, const float* __restrict__ S,
                         float* __restrict__ pooled) {
  int t = blockIdx.x * 256 + threadIdx.x;
  if (t >= 4096) return;
  int b = t >> 8, co = t & 255;
  const float* sb = S + b * 9 * 256;
  float acc = 0.f;
  for (int ci = 0; ci < 256; ++ci) {
    const float* wr = chw + ((size_t)(co * 256 + ci)) * 9;
    #pragma unroll
    for (int tap = 0; tap < 9; ++tap) acc += wr[tap] * sb[tap * 256 + ci];
  }
  pooled[t] = acc;
}

// --- ct1 -> LayerNorm(32) -> relu -> ct2 : produces channel gate cw --------
__global__ void cw_k(const float* __restrict__ pooled,
                     const float* __restrict__ ct1w, const float* __restrict__ ct1b,
                     const float* __restrict__ lng, const float* __restrict__ lnb,
                     const float* __restrict__ ct2w, const float* __restrict__ ct2b,
                     float* __restrict__ cw) {
  __shared__ float tsh[32];
  __shared__ float stats[2];
  const int b = blockIdx.x, tid = threadIdx.x;
  if (tid < 32) {
    float a = ct1b[tid];
    for (int c = 0; c < 256; ++c) a += ct1w[tid * 256 + c] * pooled[b * 256 + c];
    tsh[tid] = a;
  }
  __syncthreads();
  if (tid == 0) {
    float mu = 0.f;
    for (int j = 0; j < 32; ++j) mu += tsh[j];
    mu *= (1.f / 32.f);
    float var = 0.f;
    for (int j = 0; j < 32; ++j) { float d = tsh[j] - mu; var += d * d; }
    var *= (1.f / 32.f);
    stats[0] = mu; stats[1] = rsqrtf(var + EPSV);
  }
  __syncthreads();
  if (tid < 32) tsh[tid] = fmaxf((tsh[tid] - stats[0]) * stats[1] * lng[tid] + lnb[tid], 0.f);
  __syncthreads();
  float a = ct2b[tid];
  #pragma unroll
  for (int j = 0; j < 32; ++j) a += ct2w[tid * 32 + j] * tsh[j];
  cw[b * 256 + tid] = a;
}

// ---------------------------------------------------------------------------
extern "C" void kernel_launch(void* const* d_in, const int* in_sizes, int n_in,
                              void* d_out, int out_size, void* d_ws, size_t ws_size,
                              hipStream_t stream) {
  (void)in_sizes; (void)n_in; (void)out_size; (void)ws_size;
  const float* x        = (const float*)d_in[0];
  const float* reduce_w = (const float*)d_in[1];
  const float* reduce_b = (const float*)d_in[2];
  const float* bn_g     = (const float*)d_in[3];
  const float* bn_b     = (const float*)d_in[4];
  const float* bn_mean  = (const float*)d_in[5];
  const float* bn_var   = (const float*)d_in[6];
  const float* conv_w   = (const float*)d_in[7];
  const float* conv1_w  = (const float*)d_in[8];
  const float* conv2_w  = (const float*)d_in[9];
  const float* conv3_w  = (const float*)d_in[10];
  const float* conv4_w  = (const float*)d_in[11];
  const float* chw      = (const float*)d_in[12];
  const float* fuse_w   = (const float*)d_in[13];
  const float* ct1_w    = (const float*)d_in[14];
  const float* ct1_b    = (const float*)d_in[15];
  const float* ln_g     = (const float*)d_in[16];
  const float* ln_b     = (const float*)d_in[17];
  const float* ct2_w    = (const float*)d_in[18];
  const float* ct2_b    = (const float*)d_in[19];
  const float* proj_w   = (const float*)d_in[20];
  float* out = (float*)d_out;

  char* ws = (char*)d_ws;
  size_t o_xn  = 0;                          // x NHWC bf16: 16*16384*256
  size_t o_r   = o_xn  + (size_t)134217728;  // r NHWC bf16: 16*16384*64
  size_t o_cat = o_r   + (size_t)33554432;   // cat NHWC bf16: 16*4096*1536
  size_t o_qf  = o_cat + (size_t)201326592;  // qkv NHWC f32
  size_t o_qb  = o_qf  + (size_t)67108864;   // qkv NHWC bf16
  size_t o_wr  = o_qb  + (size_t)33554432;   // packed weights
  size_t o_w1  = o_wr  + (size_t)32768;
  size_t o_w2  = o_w1  + (size_t)131072;
  size_t o_w3  = o_w2  + (size_t)1179648;
  size_t o_w4  = o_w3  + (size_t)3276800;
  size_t o_wf  = o_w4  + (size_t)6422528;
  size_t o_wp  = o_wf  + (size_t)7077888;    // per-batch proj fragments
  size_t o_ab  = o_wp  + (size_t)2097152;    // alpha/beta
  size_t o_lg  = o_ab  + (size_t)512;        // content logits / softmax
  size_t o_S   = o_lg  + (size_t)262144;
  size_t o_pl  = o_S   + (size_t)147456;
  size_t o_cw  = o_pl  + (size_t)16384;

  unsigned short* xn   = (unsigned short*)(ws + o_xn);
  unsigned short* rbuf = (unsigned short*)(ws + o_r);
  unsigned short* cat  = (unsigned short*)(ws + o_cat);
  float*          qf   = (float*)(ws + o_qf);
  unsigned short* qb   = (unsigned short*)(ws + o_qb);
  unsigned short* wred = (unsigned short*)(ws + o_wr);
  unsigned short* w1   = (unsigned short*)(ws + o_w1);
  unsigned short* w2   = (unsigned short*)(ws + o_w2);
  unsigned short* w3   = (unsigned short*)(ws + o_w3);
  unsigned short* w4   = (unsigned short*)(ws + o_w4);
  unsigned short* wfu  = (unsigned short*)(ws + o_wf);
  unsigned short* wpj  = (unsigned short*)(ws + o_wp);
  float*          ab   = (float*)(ws + o_ab);
  float*          lg   = (float*)(ws + o_lg);
  float*          Sbuf = (float*)(ws + o_S);
  float*          pl   = (float*)(ws + o_pl);
  float*          cwb  = (float*)(ws + o_cw);

  // 1) weight packing + BN fold
  auto packLaunch = [&](const float* W, unsigned short* dst, int Cout, int Cin, int KS) {
    int total = (Cout >> 4) * (KS * KS * (Cin >> 5)) * 32;
    pack_w<<<(total + 255) / 256, 256, 0, stream>>>(W, dst, Cout, Cin, KS, total);
  };
  packLaunch(reduce_w, wred,  64,  256, 1);
  packLaunch(conv1_w,  w1,   256,  256, 1);
  packLaunch(conv2_w,  w2,   256,  256, 3);
  packLaunch(conv3_w,  w3,   256,  256, 5);
  packLaunch(conv4_w,  w4,   256,  256, 7);
  packLaunch(fuse_w,   wfu,  256, 1536, 3);
  bn_coef_k<<<1, 64, 0, stream>>>(bn_g, bn_b, bn_mean, bn_var, reduce_b, ab);

  // 2) layout conversion + maxpool branch
  x_nhwc_k<<<65536, 256, 0, stream>>>(x, xn);
  maxpool_k<<<65536, 256, 0, stream>>>(x, cat);

  // 3) reduce 1x1 conv + BN + ReLU (WMMA, 128x128 spatial, 256->64)
  conv_wmma<1, 128><<<2048, 256, 0, stream>>>(xn, 256, 0, 256, wred, 0L, 64,
                                              rbuf, 64, 0, nullptr, 0, ab, ab + 64);
  // 4) Haar DWT -> cat[0:256)
  dwt_k<<<16384, 256, 0, stream>>>(rbuf, cat);

  // 5) qkv branch convs (async-LDS WMMA, 64x64, 256->256) -> cat slices
  conv_lds<1, 256><<<1024, 256, 0, stream>>>(cat, 1536, 0, w1, 0L,
                                             cat, 1536, 256, nullptr, 0);
  conv_lds<3, 256><<<1024, 256, 0, stream>>>(cat, 1536, 0, w2, 0L,
                                             cat, 1536, 512, nullptr, 0);
  conv_lds<5, 256><<<1024, 256, 0, stream>>>(cat, 1536, 0, w3, 0L,
                                             cat, 1536, 768, nullptr, 0);
  conv_lds<7, 256><<<1024, 256, 0, stream>>>(cat, 1536, 0, w4, 0L,
                                             cat, 1536, 1024, nullptr, 0);

  // 6) fuse 3x3 conv over the 1536-ch concat -> qkv (f32 + bf16)
  conv_lds<3, 1536><<<1024, 256, 0, stream>>>(cat, 1536, 0, wfu, 0L,
                                              qb, 256, 0, qf, 1);

  // 7) content softmax + S-factorized channel attention
  content_k<<<8192, 256, 0, stream>>>(qf, conv_w, lg);
  softmax_k<<<16, 256, 0, stream>>>(lg);
  s_k<<<144, 256, 0, stream>>>(qf, lg, Sbuf);
  pooled_k<<<16, 256, 0, stream>>>(chw, Sbuf, pl);
  cw_k<<<16, 256, 0, stream>>>(pl, ct1_w, ct1_b, ln_g, ln_b, ct2_w, ct2_b, cwb);

  // 8) fold cw into proj weights (per batch) and final WMMA GEMM -> NCHW out
  pack_proj_k<<<256, 256, 0, stream>>>(proj_w, cwb, wpj);
  conv_lds<1, 256><<<1024, 256, 0, stream>>>(qb, 256, 0, wpj, 65536L,
                                             nullptr, 0, 0, out, 2);
}
